// GATEDClassifier_79585743995606
// MI455X (gfx1250) — matrix operand order repaired and checked by reading, not compiled, not run on hardware
//
#include <hip/hip_runtime.h>

#define NNODES   100000
#define HID      128
#define G3       (3 * HID)
#define NGRAPHS  64
#define NCLASSES 10
#define TSTEPS   5

typedef __attribute__((ext_vector_type(16))) _Float16 v16h;
typedef __attribute__((ext_vector_type(8)))  _Float16 v8h;
typedef __attribute__((ext_vector_type(8)))  float    v8f;

__device__ __forceinline__ v8f wmma_f16(v16h a, v16h b, v8f c) {
  // D = A(16x32 f16) x B(32x16 f16) + C(16x16 f32)
  return __builtin_amdgcn_wmma_f32_16x16x32_f16(false, a, false, b,
                                                (short)0, c, false, false);
}

// A-fragment: lane holds row m; half i -> k = khalf*16 + i.
// Two contiguous 16B runs -> ds_load_b128 from LDS-staged f16 tile.
__device__ __forceinline__ v16h frag_a(const _Float16* p) {
  v8h lo = *(const v8h*)(p);
  v8h hi = *(const v8h*)(p + 8);
  v16h r;
#pragma unroll
  for (int i = 0; i < 8; ++i) { r[i] = lo[i]; r[i + 8] = hi[i]; }
  return r;
}

__global__ void k_cvt_f16(const float* __restrict__ in, _Float16* __restrict__ out, int n) {
  int i = blockIdx.x * 256 + threadIdx.x;
  if (i < n) out[i] = (_Float16)in[i];
}

__global__ void k_zero_f32(float* __restrict__ p, int n) {
  int i = blockIdx.x * 256 + threadIdx.x;
  if (i < n) p[i] = 0.f;
}

__global__ void k_init_h(const float* __restrict__ m, float* __restrict__ h) {
  int i = blockIdx.x * 256 + threadIdx.x;
  if (i < NNODES * HID) h[i] = ((i & (HID - 1)) == 0) ? m[i >> 7] : 0.f;
}

// msg = h @ W_e^T + b_e   ([N,128] = [N,128] x [128,128]^T)
__global__ __launch_bounds__(256) void k_msg_gemm(
    const float* __restrict__ h, const _Float16* __restrict__ We,
    const float* __restrict__ be, float* __restrict__ msg) {
  __shared__ _Float16 As[16 * HID];
  const int row0 = blockIdx.x * 16;
  const int tid  = threadIdx.x;
#pragma unroll
  for (int i = 0; i < 8; ++i) {                 // stage 16x128 f32 -> f16
    int e = tid + i * 256;
    As[e] = (_Float16)h[(size_t)row0 * HID + e];
  }
  __syncthreads();
  const int lane = tid & 31, wave = tid >> 5;
  const int mrow = lane & 15, khalf = lane >> 4;
  const int ncol = wave * 16 + mrow;            // output column (lane n = L%16)
  v8f acc = {};
#pragma unroll
  for (int k0 = 0; k0 < HID; k0 += 32) {
    v16h a = frag_a(As + mrow * HID + k0 + khalf * 16);
    v16h b = *(const v16h*)(We + (size_t)ncol * HID + k0 + khalf * 16);
    acc = wmma_f16(a, b, acc);
  }
  const float bias = be[ncol];
#pragma unroll
  for (int r = 0; r < 8; ++r) {                 // C vgpr r -> row r + 8*khalf
    int row = row0 + khalf * 8 + r;
    msg[(size_t)row * HID + ncol] = acc[r] + bias;
  }
}

// a[dst[e]] += msg[src[e]] : one wave per edge, fp atomics resolve in L2
__global__ __launch_bounds__(256) void k_scatter(
    const float* __restrict__ msg, const int* __restrict__ src,
    const int* __restrict__ dst, float* __restrict__ a, int nedges) {
  int w = (blockIdx.x * 256 + threadIdx.x) >> 5;
  int lane = threadIdx.x & 31;
  if (w >= nedges) return;
  int s = src[w], d = dst[w];
  const float* mr = msg + (size_t)s * HID;
  float*       ar = a   + (size_t)d * HID;
#pragma unroll
  for (int i = 0; i < 4; ++i)
    __hip_atomic_fetch_add(ar + lane + 32 * i, mr[lane + 32 * i],
                           __ATOMIC_RELAXED, __HIP_MEMORY_SCOPE_AGENT);
}

// Fused GRU cell: gi = a@W_ih^T+b_ih, gh = h@W_hh^T+b_hh, gates, in-place h.
__global__ __launch_bounds__(256) void k_gru(
    const float* __restrict__ a, float* __restrict__ h,
    const _Float16* __restrict__ Wih, const float* __restrict__ bih,
    const _Float16* __restrict__ Whh, const float* __restrict__ bhh) {
  __shared__ _Float16 As[16 * HID];
  __shared__ _Float16 Hs[16 * HID];
  const int row0 = blockIdx.x * 16;
  const int tid  = threadIdx.x;
#pragma unroll
  for (int i = 0; i < 8; ++i) {
    int e = tid + i * 256;
    As[e] = (_Float16)a[(size_t)row0 * HID + e];
    Hs[e] = (_Float16)h[(size_t)row0 * HID + e];
  }
  __syncthreads();
  const int lane = tid & 31, wave = tid >> 5;
  const int mrow = lane & 15, khalf = lane >> 4;
  const int c = wave * 16 + mrow;               // hidden column for this tile
  v8f ir = {}, iz = {}, inn = {}, hr = {}, hz = {}, hn = {};
#pragma unroll
  for (int k0 = 0; k0 < HID; k0 += 32) {
    v16h af = frag_a(As + mrow * HID + k0 + khalf * 16);
    v16h hf = frag_a(Hs + mrow * HID + k0 + khalf * 16);
    v16h b;
    b = *(const v16h*)(Wih + (size_t)(c          ) * HID + k0 + khalf * 16); ir  = wmma_f16(af, b, ir);
    b = *(const v16h*)(Wih + (size_t)(c +     HID) * HID + k0 + khalf * 16); iz  = wmma_f16(af, b, iz);
    b = *(const v16h*)(Wih + (size_t)(c + 2 * HID) * HID + k0 + khalf * 16); inn = wmma_f16(af, b, inn);
    b = *(const v16h*)(Whh + (size_t)(c          ) * HID + k0 + khalf * 16); hr  = wmma_f16(hf, b, hr);
    b = *(const v16h*)(Whh + (size_t)(c +     HID) * HID + k0 + khalf * 16); hz  = wmma_f16(hf, b, hz);
    b = *(const v16h*)(Whh + (size_t)(c + 2 * HID) * HID + k0 + khalf * 16); hn  = wmma_f16(hf, b, hn);
  }
  const float bir = bih[c], biz = bih[c + HID], bin = bih[c + 2 * HID];
  const float bhr = bhh[c], bhz = bhh[c + HID], bhn = bhh[c + 2 * HID];
  float hnew[8];
#pragma unroll
  for (int r = 0; r < 8; ++r) {
    int row = row0 + khalf * 8 + r;
    float rg = 1.f / (1.f + __expf(-(ir[r] + bir + hr[r] + bhr)));
    float zg = 1.f / (1.f + __expf(-(iz[r] + biz + hz[r] + bhz)));
    float ng = tanhf(inn[r] + bin + rg * (hn[r] + bhn));
    float ho = h[(size_t)row * HID + c];        // read f32 h before barrier
    hnew[r] = (1.f - zg) * ng + zg * ho;
  }
  __syncthreads();                              // all reads of h tile done
#pragma unroll
  for (int r = 0; r < 8; ++r) {
    int row = row0 + khalf * 8 + r;
    h[(size_t)row * HID + c] = hnew[r];
  }
}

// Per-graph sum of relu(h); graph_ids sorted -> run-length local accumulate.
__global__ __launch_bounds__(128) void k_pool(
    const float* __restrict__ h, const int* __restrict__ gid,
    float* __restrict__ sums) {
  const int c  = threadIdx.x;                   // 0..127
  const int n0 = blockIdx.x * 32;
  int g = gid[n0];
  float acc = 0.f;
  for (int i = 0; i < 32; ++i) {
    int node = n0 + i;
    int gn = gid[node];
    if (gn != g) {
      __hip_atomic_fetch_add(&sums[g * HID + c], acc,
                             __ATOMIC_RELAXED, __HIP_MEMORY_SCOPE_AGENT);
      acc = 0.f; g = gn;
    }
    float v = h[(size_t)node * HID + c];
    acc += v > 0.f ? v : 0.f;
  }
  __hip_atomic_fetch_add(&sums[g * HID + c], acc,
                         __ATOMIC_RELAXED, __HIP_MEMORY_SCOPE_AGENT);
}

__global__ void k_count(const int* __restrict__ gid, float* __restrict__ cnt, int n) {
  int i = blockIdx.x * 256 + threadIdx.x;
  if (i < n)
    __hip_atomic_fetch_add(&cnt[gid[i]], 1.f,
                           __ATOMIC_RELAXED, __HIP_MEMORY_SCOPE_AGENT);
}

// out[g,cl] = (sums[g]/cnt[g]) . W_cls[cl] + b_cls[cl]   (tiny: 640 dots)
__global__ void k_cls(const float* __restrict__ sums, const float* __restrict__ cnt,
                      const float* __restrict__ Wc, const float* __restrict__ bc,
                      float* __restrict__ out) {
  int idx = blockIdx.x * 64 + threadIdx.x;
  if (idx >= NGRAPHS * NCLASSES) return;
  int g = idx / NCLASSES, cl = idx % NCLASSES;
  float inv = 1.f / fmaxf(cnt[g], 1.f);
  float s = bc[cl];
#pragma unroll 4
  for (int j = 0; j < HID; ++j)
    s += (sums[g * HID + j] * inv) * Wc[cl * HID + j];
  out[idx] = s;
}

extern "C" void kernel_launch(void* const* d_in, const int* in_sizes, int n_in,
                              void* d_out, int out_size, void* d_ws, size_t ws_size,
                              hipStream_t stream) {
  (void)n_in; (void)out_size; (void)ws_size;
  const float* m   = (const float*)d_in[0];
  const float* We  = (const float*)d_in[1];
  const float* be  = (const float*)d_in[2];
  const float* Wih = (const float*)d_in[3];
  const float* bih = (const float*)d_in[4];
  const float* Whh = (const float*)d_in[5];
  const float* bhh = (const float*)d_in[6];
  const float* Wc  = (const float*)d_in[7];
  const float* bc  = (const float*)d_in[8];
  const int*   src = (const int*)d_in[9];
  const int*   dst = (const int*)d_in[10];
  const int*   gid = (const int*)d_in[11];
  const int nedges = in_sizes[9];

  // Workspace carve-out (~147 MiB total; h/msg/a all L2-resident on MI455X)
  char* ws = (char*)d_ws;
  size_t off = 0;
  auto carve = [&](size_t bytes) -> void* {
    void* p = ws + off;
    off = (off + bytes + 255) & ~(size_t)255;
    return p;
  };
  float*    h     = (float*)carve((size_t)NNODES * HID * 4);
  float*    msg   = (float*)carve((size_t)NNODES * HID * 4);
  float*    acc   = (float*)carve((size_t)NNODES * HID * 4);
  _Float16* We16  = (_Float16*)carve((size_t)HID * HID * 2);
  _Float16* Wih16 = (_Float16*)carve((size_t)G3 * HID * 2);
  _Float16* Whh16 = (_Float16*)carve((size_t)G3 * HID * 2);
  float*    sums  = (float*)carve((size_t)NGRAPHS * HID * 4);
  float*    cnt   = (float*)carve((size_t)NGRAPHS * 4);

  k_cvt_f16<<<(HID * HID + 255) / 256, 256, 0, stream>>>(We, We16, HID * HID);
  k_cvt_f16<<<(G3 * HID + 255) / 256, 256, 0, stream>>>(Wih, Wih16, G3 * HID);
  k_cvt_f16<<<(G3 * HID + 255) / 256, 256, 0, stream>>>(Whh, Whh16, G3 * HID);
  k_init_h<<<(NNODES * HID + 255) / 256, 256, 0, stream>>>(m, h);

  for (int t = 0; t < TSTEPS; ++t) {
    k_msg_gemm<<<NNODES / 16, 256, 0, stream>>>(h, We16, be, msg);
    k_zero_f32<<<(NNODES * HID + 255) / 256, 256, 0, stream>>>(acc, NNODES * HID);
    k_scatter<<<(nedges + 7) / 8, 256, 0, stream>>>(msg, src, dst, acc, nedges);
    k_gru<<<NNODES / 16, 256, 0, stream>>>(acc, h, Wih16, bih, Whh16, bhh);
  }

  k_zero_f32<<<(NGRAPHS * HID + 255) / 256, 256, 0, stream>>>(sums, NGRAPHS * HID);
  k_zero_f32<<<1, 256, 0, stream>>>(cnt, NGRAPHS);
  k_pool<<<NNODES / 32, 128, 0, stream>>>(h, gid, sums);
  k_count<<<(NNODES + 255) / 256, 256, 0, stream>>>(gid, cnt, NNODES);
  k_cls<<<(NGRAPHS * NCLASSES + 63) / 64, 64, 0, stream>>>(sums, cnt, Wc, bc, (float*)d_out);
}
